// GRU_4157528343130
// MI455X (gfx1250) — compile-verified, matrix operand order checked by reading, not run on hardware
//
#include <hip/hip_runtime.h>
#include <hip/hip_bf16.h>
#include <math.h>

// Problem constants (match reference): B,L,E,H
#define B_ 64
#define L_ 512
#define E_ 512
#define H_ 512

// LDS row stride (shorts) for staged weights: 512 + 8 pad -> lane l16 starts at
// bank 4*l16 (mod 64): conflict-free within a 16-lane half.
#define WROW_ 520

typedef __attribute__((ext_vector_type(16))) __bf16        bf16x16;
typedef __attribute__((ext_vector_type(8)))  float         f32x8;
typedef __attribute__((ext_vector_type(4)))  float         f32x4;
typedef __attribute__((ext_vector_type(4)))  unsigned int  u32x4;

// ---------------------------------------------------------------------------
// Fragment helpers (CDNA5 WMMA 16x16x32 bf16, wave32)
// A (16x32 MxK): lane m = lane%16; half = lane/16,
//   elements 0..7  -> K = k0..k0+7      (k0 = kt*32 + half*8)
//   elements 8..15 -> K = k0+16..k0+23
// B (32x16 KxN): lane n = lane%16, same per-lane K pattern.
// C/D (16x16 f32): VGPR r -> M = r + 8*(lane/16); N = lane%16.
// ---------------------------------------------------------------------------

__device__ __forceinline__ u32x4 ldq(const unsigned short* p) {
  return *reinterpret_cast<const u32x4*>(p);
}

__device__ __forceinline__ bf16x16 ld_frag_bf(const unsigned short* rowp, int k0) {
  union { bf16x16 v; u32x4 q[2]; } u;
  u.q[0] = ldq(rowp + k0);
  u.q[1] = ldq(rowp + k0 + 16);
  return u.v;
}

__device__ __forceinline__ bf16x16 make_frag(u32x4 q0, u32x4 q1) {
  union { bf16x16 v; u32x4 q[2]; } u;
  u.q[0] = q0;
  u.q[1] = q1;
  return u.v;
}

__device__ __forceinline__ f32x8 wmma_bf16(bf16x16 a, bf16x16 b, f32x8 c) {
  // (neg_a, A, neg_b, B, c_mod, C, reuse_a, reuse_b)
  return __builtin_amdgcn_wmma_f32_16x16x32_bf16(false, a, false, b, (short)0, c,
                                                 false, false);
}

__device__ __forceinline__ unsigned short f32_to_bf16_bits(float f) {
  union { __bf16 h; unsigned short s; } u;
  u.h = (__bf16)f;
  return u.s;
}

__device__ __forceinline__ float sigmoidf_(float x) {
  return 1.0f / (1.0f + __expf(-x));
}

// ---------------------------------------------------------------------------
// Kernel 0: reset grid-barrier counter (must be deterministic per launch)
// ---------------------------------------------------------------------------
__global__ void gru_init_kernel(unsigned* cnt) {
  if (threadIdx.x == 0 && blockIdx.x == 0) *cnt = 0u;
}

// ---------------------------------------------------------------------------
// Kernel 1: f32 -> bf16 conversion (weights and activations)
// ---------------------------------------------------------------------------
__global__ void gru_cvt_bf16_kernel(const float* __restrict__ in,
                                    unsigned short* __restrict__ out, int n) {
  int i = blockIdx.x * blockDim.x + threadIdx.x;
  if (i < n) out[i] = f32_to_bf16_bits(in[i]);
}

// ---------------------------------------------------------------------------
// Kernel 2: projection GEMM  proj_g[(t*B+b)*H + n] = X[b,t,:] . Wg[n,:] + bg[n]
//   M = L*B = 32768 (m = t*B + b), N = H = 512, K = E = 512, 3 gates per wave.
//   One wave per 16x16 (M,N) tile -> 65536 waves = 8192 blocks x 8 waves.
//   A comes from pre-converted bf16 X; all 4 fragment streams (A + 3xB) are
//   software-pipelined one K-step ahead so no WMMA waits on a just-issued load.
// ---------------------------------------------------------------------------
__global__ void __launch_bounds__(256)
gru_proj_kernel(const unsigned short* __restrict__ Xbf,  // B*L*E bf16
                const unsigned short* __restrict__ Wx,   // 3 * H*E bf16
                const float* __restrict__ bfx,
                const float* __restrict__ bix,
                const float* __restrict__ bcx,
                float* __restrict__ proj) {              // 3 * L*B*H f32
  const int wave = blockIdx.x * (blockDim.x >> 5) + (threadIdx.x >> 5);
  const int lane = threadIdx.x & 31;
  const int half = lane >> 4;
  const int l16  = lane & 15;
  const int mt = wave >> 5;          // 0..2047
  const int nt = wave & 31;          // 0..31

  const int mrow = mt * 16 + l16;    // A row (m = t*B + b)
  const int b = mrow & (B_ - 1);
  const int t = mrow >> 6;           // /B_
  const unsigned short* xrow =
      Xbf + (size_t)b * ((size_t)L_ * E_) + (size_t)t * E_;

  const int n = nt * 16 + l16;       // B column (W row)
  const size_t HE = (size_t)H_ * E_;
  const unsigned short* wf = Wx + 0 * HE + (size_t)n * E_;
  const unsigned short* wi = Wx + 1 * HE + (size_t)n * E_;
  const unsigned short* wc = Wx + 2 * HE + (size_t)n * E_;

  // 2-slot pipelined fragment buffers: [slot][lo/hi 16B]
  u32x4 xa[2][2], fa[2][2], ia[2][2], ca[2][2];
  {
    const int k0 = half * 8;
    xa[0][0] = ldq(xrow + k0); xa[0][1] = ldq(xrow + k0 + 16);
    fa[0][0] = ldq(wf + k0);   fa[0][1] = ldq(wf + k0 + 16);
    ia[0][0] = ldq(wi + k0);   ia[0][1] = ldq(wi + k0 + 16);
    ca[0][0] = ldq(wc + k0);   ca[0][1] = ldq(wc + k0 + 16);
  }

  f32x8 accf = {}, acci = {}, accc = {};
#pragma unroll
  for (int kt = 0; kt < E_ / 32; ++kt) {
    const int s  = kt & 1;
    const int ns = s ^ 1;
    if (kt + 1 < E_ / 32) {
      const int kp = (kt + 1) * 32 + half * 8;
      xa[ns][0] = ldq(xrow + kp); xa[ns][1] = ldq(xrow + kp + 16);
      fa[ns][0] = ldq(wf + kp);   fa[ns][1] = ldq(wf + kp + 16);
      ia[ns][0] = ldq(wi + kp);   ia[ns][1] = ldq(wi + kp + 16);
      ca[ns][0] = ldq(wc + kp);   ca[ns][1] = ldq(wc + kp + 16);
    }
    const bf16x16 a = make_frag(xa[s][0], xa[s][1]);
    accf = wmma_bf16(a, make_frag(fa[s][0], fa[s][1]), accf);
    acci = wmma_bf16(a, make_frag(ia[s][0], ia[s][1]), acci);
    accc = wmma_bf16(a, make_frag(ca[s][0], ca[s][1]), accc);
  }

  const float bsf = bfx[n], bsi = bix[n], bsc = bcx[n];
  const size_t LBH = (size_t)L_ * B_ * H_;
#pragma unroll
  for (int r = 0; r < 8; ++r) {
    const int m = mt * 16 + r + 8 * half;
    const size_t idx = (size_t)m * H_ + n;
    proj[idx]           = accf[r] + bsf;
    proj[LBH + idx]     = acci[r] + bsi;
    proj[2 * LBH + idx] = accc[r] + bsc;
  }
}

// ---------------------------------------------------------------------------
// Kernel 3: persistent sequential scan.
//   32 blocks x 128 threads; block owns nt = blockIdx.x, wave w owns mt = w.
//   - Recurrent weights (3 gates x 16 hidden cols) staged once in LDS (48 KB),
//     read via ds_load_b128 every step.
//   - h state for this wave's tile lives in REGISTERS (hp8); only the bf16
//     copy used as cross-wave WMMA A-fragments goes through global memory.
//   - proj operands for step t+1 are loaded between barrier-arrive and
//     barrier-wait, hiding HBM latency behind barrier completion.
//   - out stores are nontemporal (streaming, keep L2 for proj/h/weights).
// ---------------------------------------------------------------------------
__global__ void __launch_bounds__(128)
gru_scan_kernel(const unsigned short* __restrict__ Wm,  // 3 * H*H bf16
                const float* __restrict__ bfm,
                const float* __restrict__ bim,
                const float* __restrict__ bcm,
                const float* __restrict__ proj,         // 3 * L*B*H f32
                float* __restrict__ out,                // B*L*H + B*H f32
                unsigned short* __restrict__ hb,        // 2 * B*H bf16
                unsigned* __restrict__ cnt) {
  __shared__ unsigned short Wlds[3 * 16 * WROW_];       // ~49.9 KB

  const int tid  = threadIdx.x;                         // 0..127
  const int lane = tid & 31;
  const int half = lane >> 4;
  const int l16  = lane & 15;
  const int mt   = tid >> 5;                            // 0..3  (batch tile)
  const int nt   = blockIdx.x;                          // 0..31 (hidden tile)

  const int n    = nt * 16 + l16;                       // hidden column
  const int arow = mt * 16 + l16;                       // batch row (A)

  const size_t HH  = (size_t)H_ * H_;
  const size_t BH  = (size_t)B_ * H_;
  const size_t LBH = (size_t)L_ * B_ * H_;

  // ---- stage W rows [nt*16, nt*16+16) x 3 gates into LDS (pad to WROW_) ----
  for (int c = tid; c < 3072; c += 128) {
    const int row = c >> 6;              // 0..47  (g*16 + r)
    const int k   = (c & 63) * 8;        // short offset in row
    const int g   = row >> 4;
    const int r   = row & 15;
    const unsigned short* src =
        Wm + (size_t)g * HH + (size_t)(nt * 16 + r) * H_ + k;
    *reinterpret_cast<u32x4*>(&Wlds[row * WROW_ + k]) = ldq(src);
  }
  __syncthreads();

  // per-lane LDS row pointers (B fragment source, one W row per lane)
  const unsigned short* wfl = &Wlds[(0 * 16 + l16) * WROW_];
  const unsigned short* wil = &Wlds[(1 * 16 + l16) * WROW_];
  const unsigned short* wcl = &Wlds[(2 * 16 + l16) * WROW_];

  const float bsf = bfm[n], bsi = bim[n], bsc = bcm[n];

  // h state for this tile lives in registers; preload proj for t=0.
  float fx8[8], ix8[8], cx8[8], hp8[8];
#pragma unroll
  for (int r = 0; r < 8; ++r) {
    const int b = mt * 16 + r + 8 * half;
    const size_t pidx = (size_t)b * H_ + n;   // t = 0
    fx8[r] = proj[pidx];
    ix8[r] = proj[LBH + pidx];
    cx8[r] = proj[2 * LBH + pidx];
    hp8[r] = 0.0f;
  }

  int cur = 0;
  for (int t = 0; t < L_; ++t) {
    // ---- GEMM: h(64x512) x W^T for 3 gates, A pipelined 2-deep ----
    f32x8 accf = {}, acci = {}, accc = {};
    if (t > 0) {
      const unsigned short* hrow = hb + (size_t)cur * BH + (size_t)arow * H_;
      u32x4 aq0[2], aq1[2];
      aq0[0] = ldq(hrow + 0 * 32 + half * 8);
      aq1[0] = ldq(hrow + 0 * 32 + half * 8 + 16);
      aq0[1] = ldq(hrow + 1 * 32 + half * 8);
      aq1[1] = ldq(hrow + 1 * 32 + half * 8 + 16);
#pragma unroll
      for (int kt = 0; kt < H_ / 32; ++kt) {
        const int slot = kt & 1;
        const bf16x16 a = make_frag(aq0[slot], aq1[slot]);
        if (kt + 2 < H_ / 32) {
          const int kp = (kt + 2) * 32 + half * 8;
          aq0[slot] = ldq(hrow + kp);
          aq1[slot] = ldq(hrow + kp + 16);
        }
        const int k0 = kt * 32 + half * 8;
        accf = wmma_bf16(a, ld_frag_bf(wfl, k0), accf);
        acci = wmma_bf16(a, ld_frag_bf(wil, k0), acci);
        accc = wmma_bf16(a, ld_frag_bf(wcl, k0), accc);
      }
    }

    // ---- gates + state update (h carried in registers) ----
    const int nxt = cur ^ 1;
#pragma unroll
    for (int r = 0; r < 8; ++r) {
      const int b = mt * 16 + r + 8 * half;
      const float z   = sigmoidf_(fx8[r] + accf[r] + bsf);
      const float rr  = sigmoidf_(ix8[r] + acci[r] + bsi);
      const float htl = tanhf(cx8[r] + rr * (accc[r] + bsc));
      const float hn  = (1.0f - z) * hp8[r] + z * htl;
      hp8[r] = hn;

      __builtin_nontemporal_store(
          hn, &out[(size_t)b * ((size_t)L_ * H_) + (size_t)t * H_ + n]);
      hb[(size_t)nxt * BH + (size_t)b * H_ + n] = f32_to_bf16_bits(hn);
      if (t == L_ - 1) {
        __builtin_nontemporal_store(hn, &out[LBH + (size_t)b * H_ + n]);
      }
    }

    // ---- grid barrier: arrive, then overlap t+1 proj loads with the wait ----
    __threadfence();
    __syncthreads();
    if (tid == 0) {
      __hip_atomic_fetch_add(cnt, 1u, __ATOMIC_ACQ_REL, __HIP_MEMORY_SCOPE_AGENT);
    }
    if (t + 1 < L_) {
#pragma unroll
      for (int r = 0; r < 8; ++r) {
        const int b = mt * 16 + r + 8 * half;
        const size_t pnext = ((size_t)(t + 1) * B_ + b) * H_ + n;
        fx8[r] = proj[pnext];
        ix8[r] = proj[LBH + pnext];
        cx8[r] = proj[2 * LBH + pnext];
        // warm L2 one further step ahead (global_prefetch_b8, no counters)
        if (t + 2 < L_) {
          const size_t p2 = pnext + (size_t)B_ * H_;
          __builtin_prefetch(&proj[p2], 0, 0);
          __builtin_prefetch(&proj[LBH + p2], 0, 0);
          __builtin_prefetch(&proj[2 * LBH + p2], 0, 0);
        }
      }
    }
    if (tid == 0) {
      const unsigned target = (unsigned)(t + 1) * gridDim.x;
      while (__hip_atomic_load(cnt, __ATOMIC_ACQUIRE, __HIP_MEMORY_SCOPE_AGENT) < target) {
        __builtin_amdgcn_s_sleep(1);
      }
    }
    __syncthreads();
    cur = nxt;
  }
}

// ---------------------------------------------------------------------------
// Host launcher
// ---------------------------------------------------------------------------
extern "C" void kernel_launch(void* const* d_in, const int* in_sizes, int n_in,
                              void* d_out, int out_size, void* d_ws, size_t ws_size,
                              hipStream_t stream) {
  (void)in_sizes; (void)n_in; (void)out_size; (void)ws_size;

  const float* X    = (const float*)d_in[0];
  const float* W_fx = (const float*)d_in[1];
  const float* b_fx = (const float*)d_in[2];
  const float* W_ix = (const float*)d_in[3];
  const float* b_ix = (const float*)d_in[4];
  const float* W_cx = (const float*)d_in[5];
  const float* b_cx = (const float*)d_in[6];
  const float* W_fm = (const float*)d_in[7];
  const float* b_fm = (const float*)d_in[8];
  const float* W_im = (const float*)d_in[9];
  const float* b_im = (const float*)d_in[10];
  const float* W_cm = (const float*)d_in[11];
  const float* b_cm = (const float*)d_in[12];
  float* out = (float*)d_out;

  const size_t HE   = (size_t)H_ * E_;
  const size_t HH   = (size_t)H_ * H_;
  const size_t BH   = (size_t)B_ * H_;
  const size_t LBH  = (size_t)L_ * B_ * H_;
  const size_t BLE  = (size_t)B_ * L_ * E_;

  // Workspace layout (bytes):
  //   [0,256)   : barrier counter
  //   Wx        : 3*HE bf16   (1.5 MB)
  //   Wm        : 3*HH bf16   (1.5 MB)
  //   Xbf       : BLE  bf16   (32 MB)
  //   proj      : 3*LBH f32   (192 MB)
  //   hb        : 2*BH  bf16  (128 KB)
  char* wsb = (char*)d_ws;
  unsigned*       cnt  = (unsigned*)wsb;
  unsigned short* Wx   = (unsigned short*)(wsb + 256);
  unsigned short* Wm   = Wx + 3 * HE;
  unsigned short* Xbf  = Wm + 3 * HH;
  float*          proj = (float*)(Xbf + BLE);
  unsigned short* hb   = (unsigned short*)(proj + 3 * LBH);

  // 0) reset barrier counter (deterministic across replays)
  gru_init_kernel<<<1, 64, 0, stream>>>(cnt);

  // 1) convert the six weight matrices and the input tensor to bf16
  const int nW = (int)HE;  // all weights are 512x512
  const int cvtBlocks = (nW + 255) / 256;
  gru_cvt_bf16_kernel<<<cvtBlocks, 256, 0, stream>>>(W_fx, Wx + 0 * HE, nW);
  gru_cvt_bf16_kernel<<<cvtBlocks, 256, 0, stream>>>(W_ix, Wx + 1 * HE, nW);
  gru_cvt_bf16_kernel<<<cvtBlocks, 256, 0, stream>>>(W_cx, Wx + 2 * HE, nW);
  gru_cvt_bf16_kernel<<<cvtBlocks, 256, 0, stream>>>(W_fm, Wm + 0 * HH, nW);
  gru_cvt_bf16_kernel<<<cvtBlocks, 256, 0, stream>>>(W_im, Wm + 1 * HH, nW);
  gru_cvt_bf16_kernel<<<cvtBlocks, 256, 0, stream>>>(W_cm, Wm + 2 * HH, nW);
  const int nX = (int)BLE;  // 16.7M elements
  gru_cvt_bf16_kernel<<<(nX + 255) / 256, 256, 0, stream>>>(X, Xbf, nX);

  // 2) input projections: 2048 M-tiles x 32 N-tiles = 65536 waves, 8 waves/block
  gru_proj_kernel<<<8192, 256, 0, stream>>>(Xbf, Wx, b_fx, b_ix, b_cx, proj);

  // 3) persistent recurrent scan: 32 blocks (one nt each) x 4 waves (mt)
  gru_scan_kernel<<<32, 128, 0, stream>>>(Wm, b_fm, b_im, b_cm, proj, out, hb, cnt);
}
